// EpisodicMemory_69741678953012
// MI455X (gfx1250) — matrix operand order repaired and checked by reading, not compiled, not run on hardware
//
#include <hip/hip_runtime.h>
#include <hip/hip_bf16.h>
#include <stdint.h>

// Problem constants (match reference setup_inputs)
#define HDIM   1024
#define NSTORE 65536
#define BQ     2048
#define TOPKN  8
#define NCHUNK 4096
#define NCHUNKS (NSTORE / NCHUNK)   // 16
#define SIM_ROWS 128                // query rows per workgroup in sim kernel
#define KBLK    128                 // K-slice staged in LDS per step
#define NKB     (HDIM / KBLK)       // 8
#define SWEEPS  (NCHUNK / 64)       // 64

typedef __attribute__((ext_vector_type(16))) __bf16         v16bf;
typedef __attribute__((ext_vector_type(16))) unsigned short v16us;
typedef __attribute__((ext_vector_type(8)))  unsigned short v8us;
typedef __attribute__((ext_vector_type(8)))  float          v8f;

union BFVec { v16bf bf; v16us us; };

__device__ __forceinline__ unsigned short f32_to_bf16_bits(float x) {
    union { float f; uint32_t u; } c; c.f = x;
    uint32_t u = c.u;
    u += 0x7FFFu + ((u >> 16) & 1u);   // round-to-nearest-even
    return (unsigned short)(u >> 16);
}

// ---- WMMA fragment loaders (wave32, 16x16x32 bf16), pure-bf16 sources ----
// A (16x32 MxK): lane L -> row m=L&15; K contiguous [16*(L>>4) .. +15]
__device__ __forceinline__ v16bf load_a_bf16(const unsigned short* src, int ld, int kBase, int lane) {
    int m  = lane & 15;
    int k0 = kBase + ((lane >> 4) << 4);
    const unsigned short* p = src + (size_t)m * ld + k0;
    v8us a = *(const v8us*)p;
    v8us b = *(const v8us*)(p + 8);
    BFVec r;
#pragma unroll
    for (int j = 0; j < 8; ++j) { r.us[j] = a[j]; r.us[j + 8] = b[j]; }
    return r.bf;
}
// B (32x16 KxN), source stores operand transposed row-major:
// B[k][n] = src[n*ld + k] -> 16 contiguous bf16 per lane
__device__ __forceinline__ v16bf load_b_bf16(const unsigned short* src, int ld, int kBase, int lane) {
    int n  = lane & 15;
    int k0 = kBase + ((lane >> 4) << 4);
    const unsigned short* p = src + (size_t)n * ld + k0;
    v8us a = *(const v8us*)p;
    v8us b = *(const v8us*)(p + 8);
    BFVec r;
#pragma unroll
    for (int j = 0; j < 8; ++j) { r.us[j] = a[j]; r.us[j + 8] = b[j]; }
    return r.bf;
}
// B (32x16 KxN) true row-major KxN source: B[k][n] = src[k*ld + n] (strided)
__device__ __forceinline__ v16bf load_b_cols_bf16(const unsigned short* src, int ld, int kBase, int lane) {
    int n  = lane & 15;
    int k0 = kBase + ((lane >> 4) << 4);
    BFVec r;
#pragma unroll
    for (int j = 0; j < 16; ++j) r.us[j] = src[(size_t)(k0 + j) * ld + n];
    return r.bf;
}

__device__ __forceinline__ v8f wmma_bf16(v16bf a, v16bf b, v8f c) {
    return __builtin_amdgcn_wmma_f32_16x16x32_bf16(false, a, false, b, (short)0, c, false, false);
}

// descending top-8 insertion; fully unrolled so arrays stay in VGPRs
__device__ __forceinline__ void topk_insert(float (&tv)[TOPKN], int (&ti)[TOPKN], float v, int idx) {
    if (v <= tv[TOPKN - 1]) return;
    tv[TOPKN - 1] = v; ti[TOPKN - 1] = idx;
#pragma unroll
    for (int p = TOPKN - 1; p > 0; --p) {
        if (tv[p] > tv[p - 1]) {
            float t = tv[p]; tv[p] = tv[p - 1]; tv[p - 1] = t;
            int   q = ti[p]; ti[p] = ti[p - 1]; ti[p - 1] = q;
        }
    }
}

// ---- elementwise fp32 -> bf16 materialization ----------------------------
__global__ void k_cvt_bf16(const float* __restrict__ src, unsigned short* __restrict__ dst) {
    size_t i = ((size_t)blockIdx.x * 256 + threadIdx.x) * 8;
#pragma unroll
    for (int j = 0; j < 8; ++j) dst[i + j] = f32_to_bf16_bits(src[i + j]);
}

// ---- weights + sum -------------------------------------------------------
__global__ void k_zero(float* p) { if (threadIdx.x == 0) p[0] = 0.f; }

__global__ void k_weights(const float* __restrict__ imp, const float* __restrict__ ts,
                          float* __restrict__ w, float* __restrict__ wsum) {
    int i = blockIdx.x * 256 + threadIdx.x;
    float v = 0.f;
    if (i < NSTORE) {
        float age = 1.0f - ts[i];                       // CURRENT_TS - timestamps
        float rec = __expf(-fabsf(age) * 0.01f);        // 1 - RECENCY_DECAY
        v = rec * (imp[i] + 1.0f);
        w[i] = v;
    }
    __shared__ float red[256];
    red[threadIdx.x] = v; __syncthreads();
#pragma unroll
    for (int s = 128; s > 0; s >>= 1) {
        if (threadIdx.x < s) red[threadIdx.x] += red[threadIdx.x + s];
        __syncthreads();
    }
    if (threadIdx.x == 0) atomicAdd(wsum, red[0]);
}

__global__ void k_finalize(const float* __restrict__ wsum, float* __restrict__ winv) {
    if (threadIdx.x == 0) winv[0] = 1.0f / (wsum[0] + 1e-8f);
}

// ---- query L2-normalize -> bf16 ------------------------------------------
__global__ void k_qnorm(const float* __restrict__ q, unsigned short* __restrict__ qn) {
    int b = blockIdx.x;
    const float* row = q + (size_t)b * HDIM;
    float vals[4]; float ss = 0.f;
#pragma unroll
    for (int j = 0; j < 4; ++j) { vals[j] = row[threadIdx.x + j * 256]; ss += vals[j] * vals[j]; }
    __shared__ float red[256];
    red[threadIdx.x] = ss; __syncthreads();
#pragma unroll
    for (int s = 128; s > 0; s >>= 1) {
        if (threadIdx.x < s) red[threadIdx.x] += red[threadIdx.x + s];
        __syncthreads();
    }
    __shared__ float inv;
    if (threadIdx.x == 0) inv = 1.0f / fmaxf(sqrtf(red[0]), 1e-12f);
    __syncthreads();
#pragma unroll
    for (int j = 0; j < 4; ++j)
        qn[(size_t)b * HDIM + threadIdx.x + j * 256] = f32_to_bf16_bits(vals[j] * inv);
}

// ---- WMMA keys = store@Wk^T, reduce row norms, emit per-key scale --------
__global__ __launch_bounds__(256) void k_keyscale(
        const unsigned short* __restrict__ storeb, const unsigned short* __restrict__ Wkb,
        const float* __restrict__ w, const float* __restrict__ winv,
        float* __restrict__ kscale) {
    int n0 = blockIdx.x * 16;
    int wave = threadIdx.x >> 5, lane = threadIdx.x & 31;
    const unsigned short* A = storeb + (size_t)n0 * HDIM;
    v8f acc[8] = {};
    for (int k = 0; k < HDIM; k += 32) {
        v16bf a = load_a_bf16(A, HDIM, k, lane);
#pragma unroll
        for (int t = 0; t < 8; ++t) {
            const unsigned short* Brow = Wkb + (size_t)(wave * 128 + t * 16) * HDIM;
            acc[t] = wmma_bf16(a, load_b_bf16(Brow, HDIM, k, lane), acc[t]);
        }
    }
    __shared__ float norm2[16];
    if (threadIdx.x < 16) norm2[threadIdx.x] = 0.f;
    __syncthreads();
#pragma unroll
    for (int v = 0; v < 8; ++v) {                 // C layout: VGPR v -> row v+8*(lane>=16)
        float s = 0.f;
#pragma unroll
        for (int t = 0; t < 8; ++t) s += acc[t][v] * acc[t][v];
        atomicAdd(&norm2[v + ((lane >> 4) << 3)], s);
    }
    __syncthreads();
    if (threadIdx.x < 16) {
        int n = n0 + threadIdx.x;
        kscale[n] = w[n] * winv[0] / fmaxf(sqrtf(norm2[threadIdx.x]), 1e-12f);
    }
}

// ---- WMMA qk = q_norm @ Wk (B read column-wise) -> bf16 ------------------
__global__ __launch_bounds__(256) void k_qk(
        const unsigned short* __restrict__ qn, const unsigned short* __restrict__ Wkb,
        unsigned short* __restrict__ qk) {
    int b0 = blockIdx.x * 16;
    int wave = threadIdx.x >> 5, lane = threadIdx.x & 31;
    const unsigned short* A = qn + (size_t)b0 * HDIM;
    v8f acc[8] = {};
    for (int k = 0; k < HDIM; k += 32) {
        v16bf a = load_a_bf16(A, HDIM, k, lane);
#pragma unroll
        for (int t = 0; t < 8; ++t) {
            int cb = wave * 128 + t * 16;
            acc[t] = wmma_bf16(a, load_b_cols_bf16(Wkb + cb, HDIM, k, lane), acc[t]);
        }
    }
#pragma unroll
    for (int t = 0; t < 8; ++t) {
        int col = wave * 128 + t * 16 + (lane & 15);
#pragma unroll
        for (int v = 0; v < 8; ++v) {
            int m = v + ((lane >> 4) << 3);
            qk[(size_t)(b0 + m) * HDIM + col] = f32_to_bf16_bits(acc[t][v]);
        }
    }
}

// ---- WMMA sim + per-row top-8, B staged through LDS ----------------------
// WG = 128 query rows x 4096 keys. Wave w owns rows [w*16,+16). The shared B
// tile (64 keys x KBLK) is staged once per WG in a double-buffered LDS tile;
// prefetch of the next tile overlaps WMMA compute on the current one.
__global__ __launch_bounds__(256) void k_sim_topk(
        const unsigned short* __restrict__ qk, const unsigned short* __restrict__ storeb,
        const float* __restrict__ kscale,
        float* __restrict__ candV, int* __restrict__ candI) {
    __shared__ unsigned short Bs[2][64][KBLK + 8];   // 34.0 KB; row stride 17x16B
    __shared__ float scanBuf[8][16][17];             // 8.5 KB, per-wave private

    int tid = threadIdx.x;
    int b0 = blockIdx.x * SIM_ROWS;
    int nbase = blockIdx.y * NCHUNK;
    int wave = tid >> 5, lane = tid & 31;
    const unsigned short* A = qk + (size_t)(b0 + wave * 16) * HDIM;

    float tv[TOPKN]; int ti[TOPKN];
#pragma unroll
    for (int j = 0; j < TOPKN; ++j) { tv[j] = -1e30f; ti[j] = 0; }

    // cooperative-copy mapping: 1024 16B chunks per 64xKBLK tile, 4 per thread
    int crow[4], ckc[4];
#pragma unroll
    for (int j = 0; j < 4; ++j) { int cid = tid + j * 256; crow[j] = cid >> 4; ckc[j] = (cid & 15) * 8; }

    // prologue: (sweep 0, kblock 0) -> Bs[0]
    v8us stage[4];
#pragma unroll
    for (int j = 0; j < 4; ++j)
        stage[j] = *(const v8us*)(storeb + (size_t)(nbase + crow[j]) * HDIM + ckc[j]);
#pragma unroll
    for (int j = 0; j < 4; ++j)
        *(v8us*)&Bs[0][crow[j]][ckc[j]] = stage[j];
    __syncthreads();

    int cur = 0;
    for (int sub = 0; sub < SWEEPS; ++sub) {
        int keyBase = nbase + sub * 64;
        v8f acc[4] = {};
        for (int kb = 0; kb < NKB; ++kb) {
            // prefetch next K-block (or next sweep's first) into registers
            bool haveNext = !(sub == SWEEPS - 1 && kb == NKB - 1);
            if (haveNext) {
                int nsub = (kb == NKB - 1) ? sub + 1 : sub;
                int nkb  = (kb == NKB - 1) ? 0 : kb + 1;
                const unsigned short* src = storeb + (size_t)(nbase + nsub * 64) * HDIM + nkb * KBLK;
#pragma unroll
                for (int j = 0; j < 4; ++j)
                    stage[j] = *(const v8us*)(src + (size_t)crow[j] * HDIM + ckc[j]);
            }
            // compute on Bs[cur] while the prefetch loads are in flight
            int kg = kb * KBLK;
#pragma unroll
            for (int k = 0; k < KBLK; k += 32) {
                v16bf a = load_a_bf16(A, HDIM, kg + k, lane);
#pragma unroll
                for (int t = 0; t < 4; ++t) {
                    const unsigned short* Brow = &Bs[cur][t * 16 + (lane & 15)][k + ((lane >> 4) << 4)];
                    v8us x = *(const v8us*)Brow;
                    v8us y = *(const v8us*)(Brow + 8);
                    BFVec r;
#pragma unroll
                    for (int j = 0; j < 8; ++j) { r.us[j] = x[j]; r.us[j + 8] = y[j]; }
                    acc[t] = wmma_bf16(a, r.bf, acc[t]);
                }
            }
            __syncthreads();                       // all waves done reading Bs[cur^1] epoch
            if (haveNext) {
#pragma unroll
                for (int j = 0; j < 4; ++j)
                    *(v8us*)&Bs[cur ^ 1][crow[j]][ckc[j]] = stage[j];
            }
            __syncthreads();                       // commit visible to all waves
            cur ^= 1;
        }
        // scale + per-row top-8 (per-wave LDS region; DS ops are wave-ordered)
#pragma unroll
        for (int t = 0; t < 4; ++t) {
            int col = lane & 15;
            float sc = kscale[keyBase + t * 16 + col];
#pragma unroll
            for (int v = 0; v < 8; ++v) {
                int m = v + ((lane >> 4) << 3);
                scanBuf[wave][m][col] = acc[t][v] * sc;
            }
            __builtin_amdgcn_wave_barrier();
            if (lane < 16) {
                for (int c = 0; c < 16; ++c)
                    topk_insert(tv, ti, scanBuf[wave][lane][c], keyBase + t * 16 + c);
            }
            __builtin_amdgcn_wave_barrier();
        }
    }

    if (lane < 16) {
        int row = b0 + wave * 16 + lane;
        size_t base = ((size_t)row * NCHUNKS + blockIdx.y) * TOPKN;
#pragma unroll
        for (int j = 0; j < TOPKN; ++j) { candV[base + j] = tv[j]; candI[base + j] = ti[j]; }
    }
}

// ---- merge 128 candidates -> top-8, softmax, attn-mix of store rows ------
__global__ void k_merge_mix(const float* __restrict__ candV, const int* __restrict__ candI,
                            const float* __restrict__ store, unsigned short* __restrict__ Mmix) {
    int b = blockIdx.x;
    __shared__ float attn[TOPKN];
    __shared__ int   idx[TOPKN];
    if (threadIdx.x == 0) {
        float tv[TOPKN]; int ti[TOPKN];
#pragma unroll
        for (int j = 0; j < TOPKN; ++j) { tv[j] = -1e30f; ti[j] = 0; }
        size_t base = (size_t)b * NCHUNKS * TOPKN;
        for (int c = 0; c < NCHUNKS * TOPKN; ++c)
            topk_insert(tv, ti, candV[base + c], candI[base + c]);
        float mx = tv[0], s = 0.f, e[TOPKN];
#pragma unroll
        for (int j = 0; j < TOPKN; ++j) { e[j] = __expf(tv[j] - mx); s += e[j]; }
        float inv = 1.0f / s;
#pragma unroll
        for (int j = 0; j < TOPKN; ++j) { attn[j] = e[j] * inv; idx[j] = ti[j]; }
    }
    __syncthreads();
#pragma unroll
    for (int j = 0; j < 4; ++j) {
        int h = threadIdx.x + j * 256;
        float acc = 0.f;
#pragma unroll
        for (int k = 0; k < TOPKN; ++k)
            acc += attn[k] * store[(size_t)idx[k] * HDIM + h];
        Mmix[(size_t)b * HDIM + h] = f32_to_bf16_bits(acc);
    }
}

// ---- out = A(bf16) @ W^T (W pre-converted bf16) --------------------------
template <bool OUT_F32>
__global__ __launch_bounds__(256) void k_gemm_wt(
        const unsigned short* __restrict__ A, const unsigned short* __restrict__ Wb,
        void* __restrict__ outp) {
    int b0 = blockIdx.x * 16;
    int wave = threadIdx.x >> 5, lane = threadIdx.x & 31;
    const unsigned short* Ar = A + (size_t)b0 * HDIM;
    v8f acc[8] = {};
    for (int k = 0; k < HDIM; k += 32) {
        v16bf a = load_a_bf16(Ar, HDIM, k, lane);
#pragma unroll
        for (int t = 0; t < 8; ++t) {
            const unsigned short* Brow = Wb + (size_t)(wave * 128 + t * 16) * HDIM;
            acc[t] = wmma_bf16(a, load_b_bf16(Brow, HDIM, k, lane), acc[t]);
        }
    }
#pragma unroll
    for (int t = 0; t < 8; ++t) {
        int col = wave * 128 + t * 16 + (lane & 15);
#pragma unroll
        for (int v = 0; v < 8; ++v) {
            int m = v + ((lane >> 4) << 3);
            if (OUT_F32)
                ((float*)outp)[(size_t)(b0 + m) * HDIM + col] = acc[t][v];
            else
                ((unsigned short*)outp)[(size_t)(b0 + m) * HDIM + col] = f32_to_bf16_bits(acc[t][v]);
        }
    }
}

// ---- host-side launcher ---------------------------------------------------
extern "C" void kernel_launch(void* const* d_in, const int* in_sizes, int n_in,
                              void* d_out, int out_size, void* d_ws, size_t ws_size,
                              hipStream_t stream) {
    const float* q     = (const float*)d_in[0];   // [B,H]
    const float* store = (const float*)d_in[1];   // [N,H]
    const float* imp   = (const float*)d_in[2];   // [N]
    const float* ts    = (const float*)d_in[3];   // [N]
    const float* Wk    = (const float*)d_in[4];   // [H,H]
    const float* Wv    = (const float*)d_in[5];   // [H,H]
    const float* Wo    = (const float*)d_in[6];   // [H,H]

    char* ws = (char*)d_ws;
    size_t off = 0;
    auto alloc = [&](size_t bytes) { void* p = ws + off; off += (bytes + 255) & ~size_t(255); return p; };

    unsigned short* storeb = (unsigned short*)alloc(sizeof(unsigned short) * (size_t)NSTORE * HDIM);
    unsigned short* Wkb    = (unsigned short*)alloc(sizeof(unsigned short) * (size_t)HDIM * HDIM);
    unsigned short* Wvb    = (unsigned short*)alloc(sizeof(unsigned short) * (size_t)HDIM * HDIM);
    unsigned short* Wob    = (unsigned short*)alloc(sizeof(unsigned short) * (size_t)HDIM * HDIM);
    float*          w      = (float*)alloc(sizeof(float) * NSTORE);
    float*          wsum   = (float*)alloc(sizeof(float) * 2);
    float*          winv   = wsum + 1;
    float*          kscale = (float*)alloc(sizeof(float) * NSTORE);
    unsigned short* qn     = (unsigned short*)alloc(sizeof(unsigned short) * (size_t)BQ * HDIM);
    unsigned short* qk     = (unsigned short*)alloc(sizeof(unsigned short) * (size_t)BQ * HDIM);
    float*          candV  = (float*)alloc(sizeof(float) * (size_t)BQ * NCHUNKS * TOPKN);
    int*            candI  = (int*)alloc(sizeof(int) * (size_t)BQ * NCHUNKS * TOPKN);
    unsigned short* Mmix   = (unsigned short*)alloc(sizeof(unsigned short) * (size_t)BQ * HDIM);
    unsigned short* tmp    = (unsigned short*)alloc(sizeof(unsigned short) * (size_t)BQ * HDIM);

    // materialize bf16 operands (once; HBM-bound)
    k_cvt_bf16<<<(NSTORE * (size_t)HDIM) / (256 * 8), 256, 0, stream>>>(store, storeb);
    k_cvt_bf16<<<((size_t)HDIM * HDIM) / (256 * 8), 256, 0, stream>>>(Wk, Wkb);
    k_cvt_bf16<<<((size_t)HDIM * HDIM) / (256 * 8), 256, 0, stream>>>(Wv, Wvb);
    k_cvt_bf16<<<((size_t)HDIM * HDIM) / (256 * 8), 256, 0, stream>>>(Wo, Wob);

    k_zero<<<1, 32, 0, stream>>>(wsum);
    k_weights<<<NSTORE / 256, 256, 0, stream>>>(imp, ts, w, wsum);
    k_finalize<<<1, 32, 0, stream>>>(wsum, winv);
    k_qnorm<<<BQ, 256, 0, stream>>>(q, qn);
    k_keyscale<<<NSTORE / 16, 256, 0, stream>>>(storeb, Wkb, w, winv, kscale);
    k_qk<<<BQ / 16, 256, 0, stream>>>(qn, Wkb, qk);
    k_sim_topk<<<dim3(BQ / SIM_ROWS, NCHUNKS), 256, 0, stream>>>(qk, storeb, kscale, candV, candI);
    k_merge_mix<<<BQ, 256, 0, stream>>>(candV, candI, store, Mmix);
    k_gemm_wt<false><<<BQ / 16, 256, 0, stream>>>(Mmix, Wvb, (void*)tmp);
    k_gemm_wt<true><<<BQ / 16, 256, 0, stream>>>(tmp, Wob, d_out);
}
